// NVTXWrappedMHA_73942156968658
// MI455X (gfx1250) — matrix operand order repaired and checked by reading, not compiled, not run on hardware
//
#include <hip/hip_runtime.h>
#include <hip/hip_bf16.h>
#include <stdint.h>

// Problem constants (from reference): B=4, S=2048, D=1024, H=16, HD=64
#define B_  4
#define S_  2048
#define D_  1024
#define H_  16
#define HD_ 64
#define M_TOTAL (B_ * S_)          // 8192 rows for the projection GEMMs

typedef __attribute__((ext_vector_type(16))) __bf16 bf16x16;
typedef __attribute__((ext_vector_type(8)))  __bf16 bf16x8;
typedef __attribute__((ext_vector_type(8)))  float  f32x8;

#define F32X8_ZERO {0.f,0.f,0.f,0.f,0.f,0.f,0.f,0.f}

// ---------------------------------------------------------------------------
// WMMA helper: D = A(16x32 bf16) * B(32x16 bf16) + C(16x16 f32)
// ---------------------------------------------------------------------------
__device__ __forceinline__ f32x8 wmma_bf16(bf16x16 a, bf16x16 b, f32x8 c) {
  return __builtin_amdgcn_wmma_f32_16x16x32_bf16(
      /*neg_a=*/false, a, /*neg_b=*/false, b,
      /*c_mod=*/(short)0, c, /*reuse_a=*/false, /*reuse_b=*/false);
}

// A-fragment loader: 16x32 (MxK) bf16, rows contiguous along K with stride ld.
// ISA layout (CDNA5 ISA): lanes 0-15 -> M=lane, K={0..7,16..23};
//                          lanes 16-31 -> M=lane-16, K={8..15,24..31}.
__device__ __forceinline__ bf16x16 load_a_frag(const __bf16* tile, int ld, int lane) {
  int m  = lane & 15;
  int kb = (lane >> 4) << 3;               // 0 or 8
  const __bf16* p = tile + (size_t)m * ld;
  bf16x8 lo = *(const bf16x8*)(p + kb);
  bf16x8 hi = *(const bf16x8*)(p + 16 + kb);
  bf16x16 a;
#pragma unroll
  for (int i = 0; i < 8; ++i) { a[i] = lo[i]; a[8 + i] = hi[i]; }
  return a;
}

// B-fragment loader: 32x16 (KxN) bf16 stored N-major (B(k,n) = src[n][k]).
__device__ __forceinline__ bf16x16 load_b_frag(const __bf16* tile, int ld, int lane) {
  int n  = lane & 15;
  int kb = (lane >> 4) << 3;               // 0 or 8
  const __bf16* p = tile + (size_t)n * ld + kb;
  bf16x8 lo = *(const bf16x8*)(p);
  bf16x8 hi = *(const bf16x8*)(p + 16);
  bf16x16 b;
#pragma unroll
  for (int i = 0; i < 8; ++i) { b[i] = lo[i]; b[8 + i] = hi[i]; }
  return b;
}

// ---------------------------------------------------------------------------
// f32 -> bf16 conversion
// ---------------------------------------------------------------------------
__global__ __launch_bounds__(256) void f32_to_bf16_kernel(const float* __restrict__ in,
                                                          __bf16* __restrict__ out, int n) {
  int i = blockIdx.x * 256 + threadIdx.x;
  if (i < n) out[i] = (__bf16)in[i];
}

// ---------------------------------------------------------------------------
// GEMM: Y[m, e] = sum_d A[m, d] * W[e, d]   (A: MxK bf16, W: NxK bf16)
// Block = 8 waves (2 M x 4 N), covering 64 rows x 128 cols; each wave owns a
// 32x32 tile (4 accumulators, A-frags reused across 2 B-tiles and vice versa).
// The block's 64x32 A-slab per k-step is staged into LDS with the gfx1250
// async copy path (global_load_async_to_lds_b128 / ASYNCcnt); B comes straight
// from global (weights are L2-resident on a 192MB L2).
// ---------------------------------------------------------------------------
#define MODE_QK_HEADS 0   // bf16 out[b][h][s][hd]
#define MODE_V_TRANS  1   // bf16 out[b][h][hd][s]   (kv-contiguous for PV WMMA)
#define MODE_PLAIN_F32 2  // f32 out[m][e]

__device__ __forceinline__ void store_tile(void* out, int mode, int N,
                                           int gmBase, int geBase, f32x8 c, int lane) {
  int n     = lane & 15;
  int rbase = (lane >> 4) << 3;            // C/D layout: lanes 16-31 hold M=r+8
#pragma unroll
  for (int r = 0; r < 8; ++r) {
    int gm = gmBase + rbase + r;
    int ge = geBase + n;
    float v = c[r];
    if (mode == MODE_PLAIN_F32) {
      ((float*)out)[(size_t)gm * N + ge] = v;
    } else {
      int b = gm / S_, s = gm % S_;
      int h = ge / HD_, d = ge % HD_;
      if (mode == MODE_QK_HEADS)
        ((__bf16*)out)[(((size_t)b * H_ + h) * S_ + s) * HD_ + d] = (__bf16)v;
      else
        ((__bf16*)out)[(((size_t)b * H_ + h) * HD_ + d) * S_ + s] = (__bf16)v;
    }
  }
}

__global__ __launch_bounds__(256)
void gemm_bf16_wmma(const __bf16* __restrict__ A, const __bf16* __restrict__ W,
                    void* __restrict__ out, int M, int N, int K, int mode) {
  __shared__ __bf16 shA[64 * 32];          // 4 KB: one 64x32 A slab per k-step

  int t    = threadIdx.x;
  int lane = t & 31;
  int wave = t >> 5;
  int wm   = wave >> 2;                    // 0..1  (M sub-block)
  int wn   = wave & 3;                     // 0..3  (N sub-block)
  int mblock = blockIdx.x;                 // 64 rows
  int nblock = blockIdx.y;                 // 128 cols

  // async-copy lane assignment: 256 lanes x 16 B == 64 rows x 32 bf16
  int crow = t >> 2;
  int cseg = (t & 3) * 8;
  const __bf16* aSrc = A + ((size_t)mblock * 64 + crow) * K + cseg;
  unsigned ldsDst = (unsigned)(uintptr_t)&shA[crow * 32 + cseg];

  const __bf16* bTile0 = W + (size_t)(nblock * 8 + wn * 2 + 0) * 16 * K;
  const __bf16* bTile1 = W + (size_t)(nblock * 8 + wn * 2 + 1) * 16 * K;

  f32x8 acc00 = F32X8_ZERO, acc01 = F32X8_ZERO;
  f32x8 acc10 = F32X8_ZERO, acc11 = F32X8_ZERO;

  int m2  = lane & 15;
  int kb2 = (lane >> 4) << 3;

  for (int k0 = 0; k0 < K; k0 += 32) {
    // ---- stage A slab: gfx1250 async global->LDS copy, tracked by ASYNCcnt ----
    {
      unsigned long long gsrc = (unsigned long long)(uintptr_t)(aSrc + k0);
      asm volatile("global_load_async_to_lds_b128 %0, %1, off"
                   :: "v"(ldsDst), "v"(gsrc) : "memory");
      asm volatile("s_wait_asynccnt 0x0" ::: "memory");
    }
    __syncthreads();

    // A fragments from LDS (ds_load_b128 x2 per fragment)
    bf16x16 a0, a1;
    {
      const __bf16* p0 = &shA[(wm * 32 + m2) * 32];
      const __bf16* p1 = &shA[(wm * 32 + 16 + m2) * 32];
      bf16x8 l0 = *(const bf16x8*)(p0 + kb2);
      bf16x8 h0 = *(const bf16x8*)(p0 + 16 + kb2);
      bf16x8 l1 = *(const bf16x8*)(p1 + kb2);
      bf16x8 h1 = *(const bf16x8*)(p1 + 16 + kb2);
#pragma unroll
      for (int i = 0; i < 8; ++i) {
        a0[i] = l0[i]; a0[8 + i] = h0[i];
        a1[i] = l1[i]; a1[8 + i] = h1[i];
      }
    }

    // B fragments from global
    bf16x16 b0 = load_b_frag(bTile0 + k0, K, lane);
    bf16x16 b1 = load_b_frag(bTile1 + k0, K, lane);
    if (k0 + 128 < K) {   // software prefetch -> global_prefetch_b8
      __builtin_prefetch(bTile0 + (size_t)(lane & 15) * K + k0 + 128, 0, 3);
      __builtin_prefetch(bTile1 + (size_t)(lane & 15) * K + k0 + 128, 0, 3);
    }

    acc00 = wmma_bf16(a0, b0, acc00);
    acc01 = wmma_bf16(a0, b1, acc01);
    acc10 = wmma_bf16(a1, b0, acc10);
    acc11 = wmma_bf16(a1, b1, acc11);
    __syncthreads();      // before overwriting the LDS slab
  }

  int gmBase = mblock * 64 + wm * 32;
  int geBase = nblock * 128 + wn * 32;
  store_tile(out, mode, N, gmBase +  0, geBase +  0, acc00, lane);
  store_tile(out, mode, N, gmBase +  0, geBase + 16, acc01, lane);
  store_tile(out, mode, N, gmBase + 16, geBase +  0, acc10, lane);
  store_tile(out, mode, N, gmBase + 16, geBase + 16, acc11, lane);
}

// ---------------------------------------------------------------------------
// RoPE in-place on bf16 buffer laid out [B][H][S][HD]; pairs (2i, 2i+1).
// ---------------------------------------------------------------------------
__global__ __launch_bounds__(256)
void rope_bf16_kernel(__bf16* __restrict__ buf, const int* __restrict__ pos, int npairs) {
  int i = blockIdx.x * 256 + threadIdx.x;
  if (i >= npairs) return;
  int pr = i & 31;                         // pair index within head dim (HD/2 = 32)
  int t  = i >> 5;                         // (b*H + h)*S + s
  int s  = t % S_;
  int b  = (t / S_) / H_;
  float p   = (float)pos[b * S_ + s];
  float inv = __expf(-((float)(2 * pr) / (float)HD_) * 9.210340371976184f);
  float ang = p * inv;
  float cs = __cosf(ang), sn = __sinf(ang);
  size_t base = (size_t)t * HD_ + 2 * pr;
  float x1 = (float)buf[base], x2 = (float)buf[base + 1];
  buf[base]     = (__bf16)(x1 * cs - x2 * sn);
  buf[base + 1] = (__bf16)(x1 * sn + x2 * cs);
}

// ---------------------------------------------------------------------------
// Flash attention: one wave per (b, h, 16-query tile). Online softmax.
// Q,K: bf16 [B][H][S][HD]; V: bf16 [B][H][HD][S]; O: bf16 [B][S][D].
// ---------------------------------------------------------------------------
__global__ __launch_bounds__(256)
void flash_attn_wmma(const __bf16* __restrict__ qb, const __bf16* __restrict__ kb,
                     const __bf16* __restrict__ vt, __bf16* __restrict__ ob) {
  __shared__ __bf16 pbuf[8][16 * 32];      // per-wave P staging (D-layout -> A-layout)

  int lane = threadIdx.x & 31;
  int wave = threadIdx.x >> 5;
  int gw   = blockIdx.x * 8 + wave;        // 0 .. B*H*(S/16)-1
  int qt   = gw % (S_ / 16);
  int bh   = gw / (S_ / 16);
  int bIdx = bh / H_, hIdx = bh % H_;

  const __bf16* Q  = qb + (size_t)bh * S_ * HD_ + (size_t)qt * 16 * HD_;
  const __bf16* Kp = kb + (size_t)bh * S_ * HD_;
  const __bf16* Vp = vt + (size_t)bh * HD_ * S_;

  bf16x16 aq0 = load_a_frag(Q, HD_, lane);
  bf16x16 aq1 = load_a_frag(Q + 32, HD_, lane);

  f32x8 o0 = F32X8_ZERO, o1 = F32X8_ZERO, o2 = F32X8_ZERO, o3 = F32X8_ZERO;
  float mrow[8], lrow[8];
#pragma unroll
  for (int r = 0; r < 8; ++r) { mrow[r] = -__builtin_inff(); lrow[r] = 0.f; }

  int n  = lane & 15;
  int hi = lane >> 4;
  int qrowbase = qt * 16 + hi * 8;
  const float scale = 0.125f;              // 1/sqrt(HD=64)
  int qend = qt * 16 + 15;

  for (int k0 = 0; k0 <= qend; k0 += 32) {
    f32x8 c0 = F32X8_ZERO, c1 = F32X8_ZERO;
    c0 = wmma_bf16(aq0, load_b_frag(Kp + (size_t)k0 * HD_,             HD_, lane), c0);
    c0 = wmma_bf16(aq1, load_b_frag(Kp + (size_t)k0 * HD_ + 32,        HD_, lane), c0);
    c1 = wmma_bf16(aq0, load_b_frag(Kp + (size_t)(k0 + 16) * HD_,      HD_, lane), c1);
    c1 = wmma_bf16(aq1, load_b_frag(Kp + (size_t)(k0 + 16) * HD_ + 32, HD_, lane), c1);

    int kv0 = k0 + n, kv1 = k0 + 16 + n;
#pragma unroll
    for (int r = 0; r < 8; ++r) {
      int q = qrowbase + r;
      float s0 = (kv0 <= q) ? c0[r] * scale : -__builtin_inff();
      float s1 = (kv1 <= q) ? c1[r] * scale : -__builtin_inff();
      float cm = fmaxf(s0, s1);
#pragma unroll
      for (int m = 1; m < 16; m <<= 1) cm = fmaxf(cm, __shfl_xor(cm, m, 32));
      float nm = fmaxf(mrow[r], cm);
      float f, p0, p1;
      if (nm == -__builtin_inff()) { f = 1.f; p0 = 0.f; p1 = 0.f; }
      else { f = __expf(mrow[r] - nm); p0 = __expf(s0 - nm); p1 = __expf(s1 - nm); }
      float rs = p0 + p1;
#pragma unroll
      for (int m = 1; m < 16; m <<= 1) rs += __shfl_xor(rs, m, 32);
      mrow[r] = nm;
      lrow[r] = lrow[r] * f + rs;
      o0[r] *= f; o1[r] *= f; o2[r] *= f; o3[r] *= f;
      pbuf[wave][(r + hi * 8) * 32 + n]      = (__bf16)p0;
      pbuf[wave][(r + hi * 8) * 32 + 16 + n] = (__bf16)p1;
    }

    bf16x16 ap;
    {
      int m   = lane & 15;
      int kb2 = hi << 3;
      const __bf16* p = &pbuf[wave][m * 32];
      bf16x8 lo = *(const bf16x8*)(p + kb2);
      bf16x8 hh = *(const bf16x8*)(p + 16 + kb2);
#pragma unroll
      for (int i = 0; i < 8; ++i) { ap[i] = lo[i]; ap[8 + i] = hh[i]; }
    }

    o0 = wmma_bf16(ap, load_b_frag(Vp + (size_t)0 * 16 * S_ + k0, S_, lane), o0);
    o1 = wmma_bf16(ap, load_b_frag(Vp + (size_t)1 * 16 * S_ + k0, S_, lane), o1);
    o2 = wmma_bf16(ap, load_b_frag(Vp + (size_t)2 * 16 * S_ + k0, S_, lane), o2);
    o3 = wmma_bf16(ap, load_b_frag(Vp + (size_t)3 * 16 * S_ + k0, S_, lane), o3);
  }

#pragma unroll
  for (int r = 0; r < 8; ++r) {
    float inv = 1.0f / lrow[r];
    int s = qt * 16 + hi * 8 + r;
    size_t base = ((size_t)bIdx * S_ + s) * D_ + (size_t)hIdx * HD_ + n;
    ob[base + 0]  = (__bf16)(o0[r] * inv);
    ob[base + 16] = (__bf16)(o1[r] * inv);
    ob[base + 32] = (__bf16)(o2[r] * inv);
    ob[base + 48] = (__bf16)(o3[r] * inv);
  }
}

// ---------------------------------------------------------------------------
// Host-side orchestration
// ---------------------------------------------------------------------------
extern "C" void kernel_launch(void* const* d_in, const int* in_sizes, int n_in,
                              void* d_out, int out_size, void* d_ws, size_t ws_size,
                              hipStream_t stream) {
  (void)in_sizes; (void)n_in; (void)out_size; (void)ws_size;
  const float* x    = (const float*)d_in[0];
  const int*   posi = (const int*)  d_in[1];
  const float* Wq   = (const float*)d_in[2];
  const float* Wk   = (const float*)d_in[3];
  const float* Wv   = (const float*)d_in[4];
  const float* Wo   = (const float*)d_in[5];
  float* out = (float*)d_out;

  char* ws = (char*)d_ws;
  size_t off = 0;
  auto carve = [&](size_t bytes) -> char* {
    off = (off + 255) & ~(size_t)255;
    char* p = ws + off;
    off += bytes;
    return p;
  };
  const size_t xN = (size_t)B_ * S_ * D_;  // 8,388,608
  const size_t wN = (size_t)D_ * D_;       // 1,048,576
  __bf16* xbf = (__bf16*)carve(xN * 2);
  __bf16* wqb = (__bf16*)carve(wN * 2);
  __bf16* wkb = (__bf16*)carve(wN * 2);
  __bf16* wvb = (__bf16*)carve(wN * 2);
  __bf16* wob = (__bf16*)carve(wN * 2);
  __bf16* qbf = (__bf16*)carve(xN * 2);    // [B][H][S][HD]
  __bf16* kbf = (__bf16*)carve(xN * 2);    // [B][H][S][HD]
  __bf16* vtb = (__bf16*)carve(xN * 2);    // [B][H][HD][S]
  __bf16* obf = (__bf16*)carve(xN * 2);    // [B][S][D]

  f32_to_bf16_kernel<<<(int)((xN + 255) / 256), 256, 0, stream>>>(x,  xbf, (int)xN);
  f32_to_bf16_kernel<<<(int)((wN + 255) / 256), 256, 0, stream>>>(Wq, wqb, (int)wN);
  f32_to_bf16_kernel<<<(int)((wN + 255) / 256), 256, 0, stream>>>(Wk, wkb, (int)wN);
  f32_to_bf16_kernel<<<(int)((wN + 255) / 256), 256, 0, stream>>>(Wv, wvb, (int)wN);
  f32_to_bf16_kernel<<<(int)((wN + 255) / 256), 256, 0, stream>>>(Wo, wob, (int)wN);

  dim3 ggrid(M_TOTAL / 64, D_ / 128);
  gemm_bf16_wmma<<<ggrid, 256, 0, stream>>>(xbf, wqb, qbf, M_TOTAL, D_, D_, MODE_QK_HEADS);
  gemm_bf16_wmma<<<ggrid, 256, 0, stream>>>(xbf, wkb, kbf, M_TOTAL, D_, D_, MODE_QK_HEADS);
  gemm_bf16_wmma<<<ggrid, 256, 0, stream>>>(xbf, wvb, vtb, M_TOTAL, D_, D_, MODE_V_TRANS);

  int npairs = B_ * H_ * S_ * (HD_ / 2);   // 4,194,304
  rope_bf16_kernel<<<(npairs + 255) / 256, 256, 0, stream>>>(qbf, posi, npairs);
  rope_bf16_kernel<<<(npairs + 255) / 256, 256, 0, stream>>>(kbf, posi, npairs);

  int nwaves = B_ * H_ * (S_ / 16);        // 8192
  flash_attn_wmma<<<nwaves / 8, 256, 0, stream>>>(qbf, kbf, vtb, obf);

  gemm_bf16_wmma<<<ggrid, 256, 0, stream>>>(obf, wob, out, M_TOTAL, D_, D_, MODE_PLAIN_F32);
}
